// Model_75368086110575
// MI455X (gfx1250) — matrix operand order repaired and checked by reading, not compiled, not run on hardware
//
#include <hip/hip_runtime.h>
#include <hip/hip_bf16.h>

typedef __bf16 bf16;
typedef __attribute__((ext_vector_type(16))) __bf16 v16bf;
typedef __attribute__((ext_vector_type(8)))  float   v8f;

#define DH 96
#define NNODES (512 * 256)
#define NEDGES (NNODES * 8)

union BF16Frag {
    v16bf v;
    uint4 q[2];
    bf16  e[16];
};

// ---- WMMA fragment loaders (bf16, 16x16x32) -------------------------------
// A (16xK row-major): lanes 0-15 own row M=lane, K chunks [k0,k0+8) and
// [k0+16,k0+24); lanes 16-31 own the chunks shifted by +8.
__device__ inline v16bf load_a_frag(const bf16* A, int ldk, int m0, int k0, int lane) {
    int m = m0 + (lane & 15);
    int ksel = (lane & 16) ? 8 : 0;
    const bf16* p = A + (size_t)m * ldk + k0 + ksel;
    BF16Frag f;
    f.q[0] = *(const uint4*)(p);
    f.q[1] = *(const uint4*)(p + 16);
    return f.v;
}

// B (KxN) stored N-major (Bt[n][k]): lane owns column n=lane&15; lanes 0-15
// hold K [k0,k0+8) and [k0+16,k0+24); lanes 16-31 the chunks shifted by +8.
__device__ inline v16bf load_b_frag(const bf16* Bt, int ldk, int n0, int k0, int lane) {
    int n = n0 + (lane & 15);
    int kh = (lane & 16) ? 8 : 0;
    const bf16* p = Bt + (size_t)n * ldk + k0 + kh;
    BF16Frag f;
    f.q[0] = *(const uint4*)(p);
    f.q[1] = *(const uint4*)(p + 16);
    return f.v;
}

__device__ inline v8f wmma_bf16(v16bf a, v16bf b, v8f c) {
    return __builtin_amdgcn_wmma_f32_16x16x32_bf16(false, a, false, b, (short)0, c,
                                                   false, false);
}

// Monotonic float->uint key for atomicMax-based segment max.
__device__ inline unsigned fkey(float f) {
    unsigned u = __float_as_uint(f);
    return (u & 0x80000000u) ? ~u : (u | 0x80000000u);
}
__device__ inline float fdecode(unsigned k) {
    unsigned u = (k & 0x80000000u) ? (k ^ 0x80000000u) : ~k;
    return __uint_as_float(u);
}

// ---- Weight prep: transpose + f32->bf16, zero-padded K --------------------
__global__ void transpose_to_bf16(const float* W, bf16* Wt, int K, int Nc, int Kpad) {
    int n = blockIdx.x;
    int k = threadIdx.x;
    if (k < Kpad)
        Wt[(size_t)n * Kpad + k] = (k < K) ? (bf16)W[(size_t)k * Nc + n] : (bf16)0.0f;
}

// ---- Encode: x = tanh(embed[node_ids] @ We + be), K=300 padded to 320 -----
__global__ void encode_kernel(const int* __restrict__ node_ids,
                              const float* __restrict__ embed,
                              const bf16* __restrict__ Wet,
                              const float* __restrict__ be,
                              float* __restrict__ Xf, bf16* __restrict__ Xb) {
    int lane = threadIdx.x & 31;
    int wave = threadIdx.x >> 5;        // 0..5 -> N-tile
    int m0 = blockIdx.x * 16;
    int n0 = wave * 16;
    long base = (long)node_ids[m0 + (lane & 15)] * 300;
    int ksel = (lane & 16) ? 8 : 0;
    v8f acc = {};
    for (int ki = 0; ki < 10; ki++) {
        int k0 = ki * 32;
        BF16Frag fa;
#pragma unroll
        for (int t = 0; t < 8; t++) {
            int k = k0 + ksel + t;
            fa.e[t] = (k < 300) ? (bf16)embed[base + k] : (bf16)0.0f;
            int k2 = k + 16;
            fa.e[8 + t] = (k2 < 300) ? (bf16)embed[base + k2] : (bf16)0.0f;
        }
        v16bf fb = load_b_frag(Wet, 320, n0, k0, lane);
        acc = wmma_bf16(fa.v, fb, acc);
    }
    int col = n0 + (lane & 15);
    int rb = m0 + ((lane & 16) ? 8 : 0);
    float bv = be[col];
#pragma unroll
    for (int r = 0; r < 8; r++) {
        float v = tanhf(acc[r] + bv);
        size_t idx = (size_t)(rb + r) * DH + col;
        Xf[idx] = v;
        Xb[idx] = (bf16)v;
    }
}

// ---- Generic [N,96] x [96,96] GEMM, 4 M-tiles/wave, B frags in registers --
__global__ void gemm_n96(const bf16* __restrict__ A, const bf16* __restrict__ Bt,
                         const float* __restrict__ bias, float* __restrict__ C,
                         bf16* __restrict__ Cb, int act) {
    int lane = threadIdx.x & 31;
    int wave = threadIdx.x >> 5;
    int m0 = blockIdx.x * 64;
    int n0 = wave * 16;
    v16bf b0 = load_b_frag(Bt, DH, n0, 0, lane);
    v16bf b1 = load_b_frag(Bt, DH, n0, 32, lane);
    v16bf b2 = load_b_frag(Bt, DH, n0, 64, lane);
    int col = n0 + (lane & 15);
    float bv = bias ? bias[col] : 0.0f;
#pragma unroll
    for (int mt = 0; mt < 4; mt++) {
        int m = m0 + mt * 16;
        v8f acc = {};
        acc = wmma_bf16(load_a_frag(A, DH, m, 0, lane), b0, acc);
        acc = wmma_bf16(load_a_frag(A, DH, m, 32, lane), b1, acc);
        acc = wmma_bf16(load_a_frag(A, DH, m, 64, lane), b2, acc);
        int rb = m + ((lane & 16) ? 8 : 0);
#pragma unroll
        for (int r = 0; r < 8; r++) {
            float v = acc[r] + bv;
            if (act == 1) v = tanhf(v);
            size_t idx = (size_t)(rb + r) * DH + col;
            C[idx] = v;
            if (Cb) Cb[idx] = (bf16)v;
        }
    }
}

// ---- Attention scalars: a_s = h . att_src, a_d = h . att_dst --------------
__global__ void att_scores(const float* __restrict__ H, const float* __restrict__ ats,
                           const float* __restrict__ atd, float* __restrict__ as_,
                           float* __restrict__ ad_, int n) {
    int i = blockIdx.x * blockDim.x + threadIdx.x;
    if (i >= n) return;
    const float* hr = H + (size_t)i * DH;
    float s = 0.f, d = 0.f;
#pragma unroll 4
    for (int j = 0; j < DH; j++) {
        float hv = hr[j];
        s += hv * ats[j];
        d += hv * atd[j];
    }
    as_[i] = s;
    ad_[i] = d;
}

// ---- Attention init: a = bg (broadcast), mkey = 0, denom = 0 --------------
__global__ void init_attn(const float* __restrict__ bg, float* __restrict__ A,
                          unsigned* __restrict__ mkey, float* __restrict__ denom, int n) {
    int i = blockIdx.x * blockDim.x + threadIdx.x;
    if (i < n * DH) A[i] = bg[i % DH];
    if (i < n) {
        mkey[i] = 0u;
        denom[i] = 0.0f;
    }
}

// ---- Edge pass 1: leaky-relu logits + segment max (atomic on uint key) ----
__global__ void edge_logits(const int* __restrict__ src, const int* __restrict__ dst,
                            const float* __restrict__ as_, const float* __restrict__ ad_,
                            float* __restrict__ ebuf, unsigned* __restrict__ mkey,
                            int E, int ET) {
    int i = blockIdx.x * blockDim.x + threadIdx.x;
    if (i >= ET) return;
    int s, d;
    if (i < E) { s = src[i]; d = dst[i]; }
    else       { s = d = i - E; }                 // self loops
    float e = as_[s] + ad_[d];
    e = (e > 0.0f) ? e : 0.2f * e;
    ebuf[i] = e;
    atomicMax(&mkey[d], fkey(e));
}

// ---- Edge pass 2: exp(e - max) + segment sum ------------------------------
__global__ void edge_exp(const int* __restrict__ dst, float* __restrict__ ebuf,
                         const unsigned* __restrict__ mkey, float* __restrict__ denom,
                         int E, int ET) {
    int i = blockIdx.x * blockDim.x + threadIdx.x;
    if (i >= ET) return;
    int d = (i < E) ? dst[i] : i - E;
    float ex = expf(ebuf[i] - fdecode(mkey[d]));
    ebuf[i] = ex;
    atomicAdd(&denom[d], ex);
}

// ---- Edge pass 3: a[d] += alpha * h[s]; 3 threads/edge (32 cols each) -----
__global__ void edge_aggr(const int* __restrict__ src, const int* __restrict__ dst,
                          const float* __restrict__ ebuf, const float* __restrict__ denom,
                          const float* __restrict__ H, float* __restrict__ A,
                          int E, int ET) {
    int i = blockIdx.x * blockDim.x + threadIdx.x;
    if (i >= ET) return;
    int c0 = blockIdx.y * 32;                     // column chunk [c0, c0+32)
    int s, d;
    if (i < E) { s = src[i]; d = dst[i]; }
    else       { s = d = i - E; }
    float alpha = ebuf[i] / denom[d];
    const float* hs = H + (size_t)s * DH + c0;
    float* ad = A + (size_t)d * DH + c0;
#pragma unroll 4
    for (int j = 0; j < 32; j++) atomicAdd(&ad[j], alpha * hs[j]);
}

__global__ void f32_to_bf16(const float* __restrict__ s, bf16* __restrict__ dsts, size_t n) {
    size_t i = blockIdx.x * (size_t)blockDim.x + threadIdx.x;
    if (i < n) dsts[i] = (bf16)s[i];
}

// ---- GRU gates: 5 fused GEMMs sharing A-frags of {a, x} -------------------
__global__ void gru_gates(const bf16* __restrict__ Ab, const bf16* __restrict__ Xb,
                          const float* __restrict__ Xf,
                          const bf16* __restrict__ Wz0t, const bf16* __restrict__ Wz1t,
                          const bf16* __restrict__ Wr0t, const bf16* __restrict__ Wr1t,
                          const bf16* __restrict__ Wh0t,
                          const float* __restrict__ bz0, const float* __restrict__ bz1,
                          const float* __restrict__ br0, const float* __restrict__ br1,
                          const float* __restrict__ bh0,
                          float* __restrict__ Z, float* __restrict__ T0,
                          bf16* __restrict__ XRb) {
    int lane = threadIdx.x & 31;
    int wave = threadIdx.x >> 5;
    int m0 = blockIdx.x * 16;
    int n0 = wave * 16;
    v8f az = {}, xz = {}, ar = {}, xr = {}, ah = {};
#pragma unroll
    for (int ki = 0; ki < 3; ki++) {
        int k0 = ki * 32;
        v16bf fa = load_a_frag(Ab, DH, m0, k0, lane);
        v16bf fx = load_a_frag(Xb, DH, m0, k0, lane);
        az = wmma_bf16(fa, load_b_frag(Wz0t, DH, n0, k0, lane), az);
        xz = wmma_bf16(fx, load_b_frag(Wz1t, DH, n0, k0, lane), xz);
        ar = wmma_bf16(fa, load_b_frag(Wr0t, DH, n0, k0, lane), ar);
        xr = wmma_bf16(fx, load_b_frag(Wr1t, DH, n0, k0, lane), xr);
        ah = wmma_bf16(fa, load_b_frag(Wh0t, DH, n0, k0, lane), ah);
    }
    __syncthreads();   // XRb aliases Ab: all waves must finish reading A rows
    int col = n0 + (lane & 15);
    int rb = m0 + ((lane & 16) ? 8 : 0);
    float bzv = bz0[col] + bz1[col];
    float brv = br0[col] + br1[col];
    float bhv = bh0[col];
#pragma unroll
    for (int r = 0; r < 8; r++) {
        size_t idx = (size_t)(rb + r) * DH + col;
        float zv = 1.0f / (1.0f + expf(-(az[r] + xz[r] + bzv)));
        float rv = 1.0f / (1.0f + expf(-(ar[r] + xr[r] + brv)));
        Z[idx] = zv;
        T0[idx] = ah[r] + bhv;
        XRb[idx] = (bf16)(Xf[idx] * rv);
    }
}

// ---- GRU update: hh = tanh(t0 + (x*r)@Wh1 + bh1); x = hh*z + x*(1-z) ------
// 4 M-tiles per wave, B frags held in registers.
__global__ void gru_update(const bf16* __restrict__ XRb, const bf16* __restrict__ Wh1t,
                           const float* __restrict__ T0, const float* __restrict__ Z,
                           const float* __restrict__ bh1,
                           float* __restrict__ Xf, bf16* __restrict__ Xb) {
    int lane = threadIdx.x & 31;
    int wave = threadIdx.x >> 5;
    int m0 = blockIdx.x * 64;
    int n0 = wave * 16;
    v16bf b0 = load_b_frag(Wh1t, DH, n0, 0, lane);
    v16bf b1 = load_b_frag(Wh1t, DH, n0, 32, lane);
    v16bf b2 = load_b_frag(Wh1t, DH, n0, 64, lane);
    int col = n0 + (lane & 15);
    float bv = bh1[col];
#pragma unroll
    for (int mt = 0; mt < 4; mt++) {
        int m = m0 + mt * 16;
        v8f acc = {};
        acc = wmma_bf16(load_a_frag(XRb, DH, m, 0, lane), b0, acc);
        acc = wmma_bf16(load_a_frag(XRb, DH, m, 32, lane), b1, acc);
        acc = wmma_bf16(load_a_frag(XRb, DH, m, 64, lane), b2, acc);
        int rb = m + ((lane & 16) ? 8 : 0);
#pragma unroll
        for (int r = 0; r < 8; r++) {
            size_t idx = (size_t)(rb + r) * DH + col;
            float hh = tanhf(acc[r] + T0[idx] + bv);
            float zv = Z[idx];
            float xn = hh * zv + Xf[idx] * (1.0f - zv);
            Xf[idx] = xn;
            Xb[idx] = (bf16)xn;
        }
    }
}

// ---- Readout: per-graph max+mean pool over L=256, then @ Wmlp -------------
__global__ void pool_mlp(const float* __restrict__ Y, const float* __restrict__ Wmlp,
                         float* __restrict__ out) {
    __shared__ float pooled[DH];
    int b = blockIdx.x;
    int t = threadIdx.x;
    if (t < DH) {
        const float* yb = Y + (size_t)b * 256 * DH + t;
        float mx = -3.0e38f, sm = 0.0f;
        for (int l = 0; l < 256; l++) {
            float v = yb[(size_t)l * DH];
            mx = fmaxf(mx, v);
            sm += v;
        }
        pooled[t] = mx + sm * (1.0f / 256.0f);
    }
    __syncthreads();
    if (t < 20) {
        float acc = 0.0f;
#pragma unroll 4
        for (int j = 0; j < DH; j++) acc += pooled[j] * Wmlp[j * 20 + t];
        out[b * 20 + t] = acc;
    }
}

extern "C" void kernel_launch(void* const* d_in, const int* in_sizes, int n_in,
                              void* d_out, int out_size, void* d_ws, size_t ws_size,
                              hipStream_t stream) {
    (void)in_sizes; (void)n_in; (void)out_size; (void)ws_size;
    const int N = NNODES, E = NEDGES, ET = NEDGES + NNODES;

    const int*   node_ids = (const int*)d_in[0];
    const int*   src      = (const int*)d_in[1];
    const int*   dst      = (const int*)d_in[2];
    const float* embed    = (const float*)d_in[3];
    const float* We  = (const float*)d_in[4];
    const float* be  = (const float*)d_in[5];
    const float* Wg  = (const float*)d_in[6];
    const float* ats = (const float*)d_in[7];
    const float* atd = (const float*)d_in[8];
    const float* bg  = (const float*)d_in[9];
    const float* Wz0 = (const float*)d_in[10]; const float* bz0 = (const float*)d_in[11];
    const float* Wz1 = (const float*)d_in[12]; const float* bz1 = (const float*)d_in[13];
    const float* Wr0 = (const float*)d_in[14]; const float* br0 = (const float*)d_in[15];
    const float* Wr1 = (const float*)d_in[16]; const float* br1 = (const float*)d_in[17];
    const float* Wh0 = (const float*)d_in[18]; const float* bh0 = (const float*)d_in[19];
    const float* Wh1 = (const float*)d_in[20]; const float* bh1 = (const float*)d_in[21];
    const float* Wemb = (const float*)d_in[22];
    const float* Wmlp = (const float*)d_in[23];
    float* out = (float*)d_out;

    // ---- workspace carve-out (aliasing: Bf={h,t0,y}, Cf={a,z}, ab={a_b,xr_b})
    char* ws = (char*)d_ws;
    size_t off = 0;
    auto alloc = [&](size_t bytes) -> void* {
        void* p = ws + off;
        off += (bytes + 255) & ~(size_t)255;
        return p;
    };
    bf16* wet   = (bf16*)alloc((size_t)DH * 320 * 2);
    bf16* wgt   = (bf16*)alloc((size_t)DH * DH * 2);
    bf16* wz0t  = (bf16*)alloc((size_t)DH * DH * 2);
    bf16* wz1t  = (bf16*)alloc((size_t)DH * DH * 2);
    bf16* wr0t  = (bf16*)alloc((size_t)DH * DH * 2);
    bf16* wr1t  = (bf16*)alloc((size_t)DH * DH * 2);
    bf16* wh0t  = (bf16*)alloc((size_t)DH * DH * 2);
    bf16* wh1t  = (bf16*)alloc((size_t)DH * DH * 2);
    bf16* wembt = (bf16*)alloc((size_t)DH * DH * 2);
    float*    xf    = (float*)alloc((size_t)N * DH * 4);
    bf16*     xb    = (bf16*)alloc((size_t)N * DH * 2);
    float*    bf_   = (float*)alloc((size_t)N * DH * 4);   // h / t0 / y
    float*    cf    = (float*)alloc((size_t)N * DH * 4);   // a / z
    bf16*     ab    = (bf16*)alloc((size_t)N * DH * 2);    // a_bf16 / xr_bf16
    float*    as_   = (float*)alloc((size_t)N * 4);
    float*    ad_   = (float*)alloc((size_t)N * 4);
    unsigned* mkey  = (unsigned*)alloc((size_t)N * 4);
    float*    denom = (float*)alloc((size_t)N * 4);
    float*    ebuf  = (float*)alloc((size_t)ET * 4);

    const int mblocks16 = N / 16;        // 8192 blocks (gates)
    const int mblocks64 = N / 64;        // 2048 blocks (gemm/update, 4 tiles/wave)
    const dim3 gthreads(192);            // 6 waves -> all 96 columns
    const int eth = 256;
    const int eblocks = (ET + eth - 1) / eth;
    const int nblocks96 = (N * DH + eth - 1) / eth;

    // weights -> bf16 N-major
    transpose_to_bf16<<<DH, 320, 0, stream>>>(We, wet, 300, DH, 320);
    transpose_to_bf16<<<DH, DH, 0, stream>>>(Wg,  wgt,  DH, DH, DH);
    transpose_to_bf16<<<DH, DH, 0, stream>>>(Wz0, wz0t, DH, DH, DH);
    transpose_to_bf16<<<DH, DH, 0, stream>>>(Wz1, wz1t, DH, DH, DH);
    transpose_to_bf16<<<DH, DH, 0, stream>>>(Wr0, wr0t, DH, DH, DH);
    transpose_to_bf16<<<DH, DH, 0, stream>>>(Wr1, wr1t, DH, DH, DH);
    transpose_to_bf16<<<DH, DH, 0, stream>>>(Wh0, wh0t, DH, DH, DH);
    transpose_to_bf16<<<DH, DH, 0, stream>>>(Wh1, wh1t, DH, DH, DH);
    transpose_to_bf16<<<DH, DH, 0, stream>>>(Wemb, wembt, DH, DH, DH);

    // encode
    encode_kernel<<<mblocks16, gthreads, 0, stream>>>(node_ids, embed, wet, be, xf, xb);

    for (int step = 0; step < 2; step++) {
        // h = x @ Wg
        gemm_n96<<<mblocks64, gthreads, 0, stream>>>(xb, wgt, nullptr, bf_, nullptr, 0);
        att_scores<<<(N + eth - 1) / eth, eth, 0, stream>>>(bf_, ats, atd, as_, ad_, N);
        init_attn<<<nblocks96, eth, 0, stream>>>(bg, cf, mkey, denom, N);
        edge_logits<<<eblocks, eth, 0, stream>>>(src, dst, as_, ad_, ebuf, mkey, E, ET);
        edge_exp<<<eblocks, eth, 0, stream>>>(dst, ebuf, mkey, denom, E, ET);
        edge_aggr<<<dim3(eblocks, 3), eth, 0, stream>>>(src, dst, ebuf, denom, bf_, cf,
                                                        E, ET);
        f32_to_bf16<<<nblocks96, eth, 0, stream>>>(cf, ab, (size_t)N * DH);
        // GRU
        gru_gates<<<mblocks16, gthreads, 0, stream>>>(ab, xb, xf, wz0t, wz1t, wr0t, wr1t,
                                                      wh0t, bz0, bz1, br0, br1, bh0,
                                                      cf, bf_, ab);
        gru_update<<<mblocks64, gthreads, 0, stream>>>(ab, wh1t, bf_, cf, bh1, xf, xb);
    }

    // y = tanh(x @ Wemb), then pooled readout @ Wmlp
    gemm_n96<<<mblocks64, gthreads, 0, stream>>>(xb, wembt, nullptr, bf_, nullptr, 1);
    pool_mlp<<<512, 128, 0, stream>>>(bf_, Wmlp, out);
}
